// SelfAttention_34127810134158
// MI455X (gfx1250) — matrix operand order repaired and checked by reading, not compiled, not run on hardware
//
#include <hip/hip_runtime.h>
#include <hip/hip_bf16.h>

typedef __attribute__((ext_vector_type(16))) __bf16 v16bf;
typedef __attribute__((ext_vector_type(8)))  float  v8f;

// Problem constants (B,C,H,W = 4,512,64,64; A=512)
constexpr int Bn = 4;
constexpr int Cc = 512;          // channels == attention dim
constexpr int Nn = 4096;         // H*W
constexpr int Jj = 1536;         // Q|K|V concatenated projection width

// LDS tile row stride in bf16 elements: 32 K-elems + 8 pad = 80 bytes
// -> 16B-aligned rows for b128 LDS ops; 20-bank stride -> conflict-free
constexpr int LDSK = 40;

// ---------------------------------------------------------------- pack x^T
__global__ __launch_bounds__(256) void pack_xT_kernel(
    const float* __restrict__ x, __bf16* __restrict__ xT) {
  __shared__ float tile[32][33];
  const int b  = blockIdx.z;
  const int n0 = blockIdx.x * 32;
  const int c0 = blockIdx.y * 32;
  const int tx = threadIdx.x;      // 0..31
  const int ty = threadIdx.y;      // 0..7
  const float* xb = x + (size_t)b * Cc * Nn;
#pragma unroll
  for (int i = 0; i < 4; ++i) {
    const int c = ty + i * 8;
    tile[c][tx] = xb[(size_t)(c0 + c) * Nn + n0 + tx];
  }
  __syncthreads();
  __bf16* xTb = xT + (size_t)b * Nn * Cc;
#pragma unroll
  for (int i = 0; i < 4; ++i) {
    const int n = ty + i * 8;
    xTb[(size_t)(n0 + n) * Cc + c0 + tx] = (__bf16)tile[tx][n];
  }
}

// ------------------------------------------------------- pack [Wq|Wk|Wv]
__global__ __launch_bounds__(256) void pack_w_kernel(
    const float* __restrict__ Wq, const float* __restrict__ Wk,
    const float* __restrict__ Wv, const float* __restrict__ bq,
    const float* __restrict__ bk, const float* __restrict__ bv,
    __bf16* __restrict__ Wc, float* __restrict__ bc) {
  const int idx = blockIdx.x * 256 + threadIdx.x;   // 0 .. 1536*512-1
  const int j = idx >> 9;
  const int c = idx & 511;
  const int jr = j & 511;
  const float* src = (j < 512) ? Wq : ((j < 1024) ? Wk : Wv);
  Wc[idx] = (__bf16)src[jr * 512 + c];
  if (c == 0) bc[j] = (j < 512) ? bq[jr] : ((j < 1024) ? bk[jr] : bv[jr]);
}

// ------------------------------------------- V transpose: P[m][1024+c]->Vt[c][m]
__global__ __launch_bounds__(256) void transpose_v_kernel(
    const __bf16* __restrict__ P, __bf16* __restrict__ Vt) {
  __shared__ __bf16 tile[32][33];
  const int b  = blockIdx.z;
  const int m0 = blockIdx.x * 32;
  const int c0 = blockIdx.y * 32;
  const int tx = threadIdx.x;
  const int ty = threadIdx.y;
  const __bf16* Pb = P + (size_t)b * Nn * Jj + 2 * Cc;
#pragma unroll
  for (int i = 0; i < 4; ++i) {
    const int m = ty + i * 8;
    tile[m][tx] = Pb[(size_t)(m0 + m) * Jj + c0 + tx];
  }
  __syncthreads();
  __bf16* Vb = Vt + (size_t)b * Cc * Nn;
#pragma unroll
  for (int i = 0; i < 4; ++i) {
    const int c = ty + i * 8;
    Vb[(size_t)(c0 + c) * Nn + m0 + tx] = tile[tx][c];
  }
}

// --------------------------------------- async global->LDS tile (row-major)
// global [ROWS][K] (K contiguous) -> sm[r*LDSK + k], 16B per lane, ASYNCcnt
template <int ROWS>
static __device__ inline void async_tile_rowmajor(__bf16* sm,
                                                  const __bf16* __restrict__ g,
                                                  size_t row0, int ld, int kk,
                                                  int tid) {
#pragma unroll
  for (int h = 0; h < ROWS / 64; ++h) {
    const int rr = (tid >> 2) + h * 64;
    const int kc = (tid & 3) * 8;
    const unsigned lds_off = (unsigned)(uintptr_t)(sm + rr * LDSK + kc);
    const __bf16* gp = g + (row0 + rr) * (size_t)ld + kk + kc;
    asm volatile("global_load_async_to_lds_b128 %0, %1, off"
                 :: "v"(lds_off), "v"((unsigned long long)(uintptr_t)gp)
                 : "memory");
  }
}

template <int N>
static __device__ inline void wait_asynccnt() {
  asm volatile("s_wait_asynccnt %0" :: "i"(N) : "memory");
}

// -------------------------------------------- WMMA fragment construction
// A 16x32 bf16 (M x K): lane m = L&15; lanes>=16 see K+8 in vgpr0-3, K+24 in 4-7
static __device__ inline v16bf frag_A16(const __bf16* sm, int mbase, int lane) {
  const __bf16* row = sm + (mbase + (lane & 15)) * LDSK;
  const int h8 = (lane >> 4) * 8;
  v16bf f;
#pragma unroll
  for (int j = 0; j < 4; ++j) {
    f[2 * j]         = row[h8 + 2 * j];
    f[2 * j + 1]     = row[h8 + 2 * j + 1];
    f[8 + 2 * j]     = row[16 + h8 + 2 * j];
    f[8 + 2 * j + 1] = row[16 + h8 + 2 * j + 1];
  }
  return f;
}

// B 32x16 bf16 (K x N): lane n = L&15; lanes>=16 see K+8 in vgpr0-3, K+24 in 4-7
static __device__ inline v16bf frag_B16(const __bf16* sm, int nbase, int lane) {
  const __bf16* row = sm + (nbase + (lane & 15)) * LDSK;
  const int kh = (lane >> 4) * 8;
  v16bf f;
#pragma unroll
  for (int j = 0; j < 8; ++j) {
    f[2 * j]     = row[kh + 2 * j + (j >= 4 ? 8 : 0)];
    f[2 * j + 1] = row[kh + 2 * j + 1 + (j >= 4 ? 8 : 0)];
  }
  return f;
}

// ------------------------------------------------------------ GEMM kernel
// NT-form: C[M,N] = A[M][K] * B[N][K]^T, bf16 in, f32 WMMA accum.
//   EPI 0: bf16 out + bias   EPI 1: f32 out   EPI 2: dst = resid + 0.1*val
// Double-buffered LDS, async global->LDS copies (ASYNCcnt pipelined).
template <int BM, int BN, int EPI>
__global__ __launch_bounds__(256) void gemm_bf16_wmma(
    const __bf16* __restrict__ Ag, int lda, size_t strideAb,
    const __bf16* __restrict__ Bg, int ldb, size_t strideBb, int K,
    __bf16* __restrict__ outH, float* __restrict__ outF, int ldo,
    size_t strideOb, const float* __restrict__ bias,
    const float* __restrict__ resid, float* __restrict__ dst) {
  constexpr int WAVES_M = BM / 32;       // each wave covers 32 output rows
  constexpr int WAVES_N = 8 / WAVES_M;   // 8 wave32s per block
  constexpr int WCOLS   = BN / WAVES_N;  // columns per wave
  constexpr int WNT     = WCOLS / 16;    // 16-wide n-fragments per wave
  constexpr int ASZ     = BM * LDSK;
  constexpr int BSZ     = BN * LDSK;
  constexpr int NASYNC  = BM / 64 + BN / 64;  // async instrs per wave per stage

  __shared__ __attribute__((aligned(16))) __bf16 As[2 * ASZ];
  __shared__ __attribute__((aligned(16))) __bf16 Bs[2 * BSZ];

  const int tid  = threadIdx.x;
  const int lane = tid & 31;
  const int wave = tid >> 5;
  const int wm = (wave % WAVES_M) * 32;
  const int wn = (wave / WAVES_M) * WCOLS;
  const size_t m0 = (size_t)blockIdx.y * BM;
  const size_t n0 = (size_t)blockIdx.x * BN;
  const int b = blockIdx.z;
  Ag += (size_t)b * strideAb;
  Bg += (size_t)b * strideBb;

  v8f acc[2][WNT];
#pragma unroll
  for (int i = 0; i < 2; ++i)
#pragma unroll
    for (int j = 0; j < WNT; ++j) {
      v8f z = {0.f, 0.f, 0.f, 0.f, 0.f, 0.f, 0.f, 0.f};
      acc[i][j] = z;
    }

  // prologue: stage 0 in flight
  async_tile_rowmajor<BM>(As, Ag, m0, lda, 0, tid);
  async_tile_rowmajor<BN>(Bs, Bg, n0, ldb, 0, tid);

  const int nsteps = K / 32;
  for (int s = 0; s < nsteps; ++s) {
    const int cur = s & 1;
    const __bf16* Ac = As + cur * ASZ;
    const __bf16* Bc = Bs + cur * BSZ;
    if (s + 1 < nsteps) {
      // issue next stage into the other buffer (safe: last read of that
      // buffer was at step s-1, sealed by its trailing barrier)
      async_tile_rowmajor<BM>(As + (cur ^ 1) * ASZ, Ag, m0, lda, (s + 1) * 32, tid);
      async_tile_rowmajor<BN>(Bs + (cur ^ 1) * BSZ, Bg, n0, ldb, (s + 1) * 32, tid);
      // async completes in order: <=NASYNC outstanding => stage s landed
      wait_asynccnt<NASYNC>();
    } else {
      wait_asynccnt<0>();
    }
    __syncthreads();

    const v16bf a0 = frag_A16(Ac, wm, lane);
    const v16bf a1 = frag_A16(Ac, wm + 16, lane);
#pragma unroll
    for (int j = 0; j < WNT; ++j) {
      const v16bf bfr = frag_B16(Bc, wn + j * 16, lane);
      acc[0][j] = __builtin_amdgcn_wmma_f32_16x16x32_bf16(
          false, a0, false, bfr, (short)0, acc[0][j], false, false);
      acc[1][j] = __builtin_amdgcn_wmma_f32_16x16x32_bf16(
          false, a1, false, bfr, (short)0, acc[1][j], false, false);
    }
    __syncthreads();
  }

  // C/D layout: lane col = L&15; vgpr r holds row r (lanes<16) / row 8+r
  const int cl = lane & 15;
  const int rh = (lane >> 4) * 8;
#pragma unroll
  for (int i = 0; i < 2; ++i)
#pragma unroll
    for (int j = 0; j < WNT; ++j) {
      const size_t col  = n0 + wn + j * 16 + cl;
      const size_t rowb = m0 + wm + i * 16 + rh;
#pragma unroll
      for (int r = 0; r < 8; ++r) {
        const float v = acc[i][j][r];
        const size_t off =
            (size_t)b * strideOb + (rowb + r) * (size_t)ldo + col;
        if (EPI == 0)      outH[off] = (__bf16)(v + bias[col]);
        else if (EPI == 1) outF[off] = v;
        else               dst[off] = resid[off] + 0.1f * v;
      }
    }
}

// ------------------------------------------ column softmax over T[n][m] (n axis)
// Stage A: per 64-row segment, per column: max + sum(exp(v-max)), vals in regs
__global__ __launch_bounds__(256) void softmax_partial_kernel(
    const float* __restrict__ T, float* __restrict__ pmax,
    float* __restrict__ psum) {
  const int m = blockIdx.x * 256 + threadIdx.x;   // column
  const int seg = blockIdx.y;                      // 0..63 (64 rows each)
  const float* base = T + (size_t)(seg * 64) * Nn + m;
  float v[64];
  float mx = -3.0e38f;
#pragma unroll
  for (int r = 0; r < 64; ++r) {
    v[r] = base[(size_t)r * Nn];
    mx = fmaxf(mx, v[r]);
  }
  float s = 0.f;
#pragma unroll
  for (int r = 0; r < 64; ++r) s += __expf(v[r] - mx);
  pmax[seg * Nn + m] = mx;
  psum[seg * Nn + m] = s;
}

// Stage B: combine 64 segment partials per column
__global__ __launch_bounds__(256) void softmax_combine_kernel(
    const float* __restrict__ pmax, const float* __restrict__ psum,
    float* __restrict__ colmax, float* __restrict__ colinv) {
  const int m = blockIdx.x * 256 + threadIdx.x;
  float gm = -3.0e38f;
#pragma unroll 8
  for (int s = 0; s < 64; ++s) gm = fmaxf(gm, pmax[s * Nn + m]);
  float gs = 0.f;
#pragma unroll 8
  for (int s = 0; s < 64; ++s)
    gs += psum[s * Nn + m] * __expf(pmax[s * Nn + m] - gm);
  colmax[m] = gm;
  colinv[m] = 1.0f / gs;
}

// Stage C: normalize + cast to bf16, 8 elems/thread (one row chunk)
__global__ __launch_bounds__(256) void softmax_norm_kernel(
    const float* __restrict__ T, const float* __restrict__ colmax,
    const float* __restrict__ colinv, __bf16* __restrict__ Tsm) {
  const size_t i0 = ((size_t)blockIdx.x * 256 + threadIdx.x) * 8;
  const int m = (int)(i0 & (size_t)(Nn - 1));
  const float4 a = *(const float4*)(T + i0);
  const float4 b = *(const float4*)(T + i0 + 4);
  const float vv[8] = {a.x, a.y, a.z, a.w, b.x, b.y, b.z, b.w};
  __bf16 o[8];
#pragma unroll
  for (int j = 0; j < 8; ++j)
    o[j] = (__bf16)(__expf(vv[j] - colmax[m + j]) * colinv[m + j]);
  *(uint4*)(Tsm + i0) = *(const uint4*)o;
}

// ---------------------------------------------------------------- launcher
extern "C" void kernel_launch(void* const* d_in, const int* in_sizes, int n_in,
                              void* d_out, int out_size, void* d_ws,
                              size_t ws_size, hipStream_t stream) {
  const float* x  = (const float*)d_in[0];
  const float* Wq = (const float*)d_in[1];
  const float* bq = (const float*)d_in[2];
  const float* Wk = (const float*)d_in[3];
  const float* bk = (const float*)d_in[4];
  const float* Wv = (const float*)d_in[5];
  const float* bv = (const float*)d_in[6];
  float* out = (float*)d_out;

  // workspace layout (bytes, 256-aligned); total ~180 MB
  char* ws = (char*)d_ws;
  __bf16* xT   = (__bf16*)(ws + 0);            // 4*4096*512*2   = 16 MB
  __bf16* Wc   = (__bf16*)(ws + 16777216);     // 1536*512*2     = 1.5 MB
  float*  bc   = (float*)(ws + 18350080);      // 1536*4
  __bf16* P    = (__bf16*)(ws + 18356224);     // 4*4096*1536*2  = 48 MB
  __bf16* Vt   = (__bf16*)(ws + 68687872);     // 4*512*4096*2   = 16 MB
  float*  Tf   = (float*)(ws + 85465088);      // 4096*4096*4    = 64 MB (per-batch reuse)
  __bf16* Tsm  = (__bf16*)(ws + 152573952);    // 4096*4096*2    = 32 MB (per-batch reuse)
  float*  pmax = (float*)(ws + 186128384);     // 64*4096*4      = 1 MB
  float*  psum = (float*)(ws + 187176960);     // 64*4096*4      = 1 MB
  float*  cmx  = (float*)(ws + 188225536);     // 4096*4
  float*  cin  = (float*)(ws + 188241920);     // 4096*4

  // 1) x -> x^T bf16 ; weights -> bf16 concat
  pack_xT_kernel<<<dim3(Nn / 32, Cc / 32, Bn), dim3(32, 8), 0, stream>>>(x, xT);
  pack_w_kernel<<<(Jj * Cc) / 256, 256, 0, stream>>>(Wq, Wk, Wv, bq, bk, bv,
                                                     Wc, bc);

  // 2) fused projections: P[b,n,0:1536] = xT[b]·Wc^T + bc   (all batches)
  gemm_bf16_wmma<128, 128, 0><<<dim3(Jj / 128, Nn / 128, Bn), 256, 0, stream>>>(
      xT, Cc, (size_t)Nn * Cc, Wc, Cc, 0, Cc, P, nullptr, Jj, (size_t)Nn * Jj,
      bc, nullptr, nullptr);

  // 2b) V -> Vt[c][m] so the PV GEMM is NT-form too
  transpose_v_kernel<<<dim3(Nn / 32, Cc / 32, Bn), dim3(32, 8), 0, stream>>>(
      P, Vt);

  // 3) per batch: scores, column softmax (axis=-2), PV + fused residual
  for (int b = 0; b < Bn; ++b) {
    const __bf16* Pb  = P + (size_t)b * Nn * Jj;
    const __bf16* Vtb = Vt + (size_t)b * Cc * Nn;
    // T[n,m] = sum_a Q[n,a]*K[m,a] in f32
    gemm_bf16_wmma<128, 128, 1><<<dim3(Nn / 128, Nn / 128, 1), 256, 0, stream>>>(
        Pb, Jj, 0, Pb + Cc, Jj, 0, Cc, nullptr, Tf, Nn, 0, nullptr, nullptr,
        nullptr);
    // softmax down columns of T (over queries n), probs -> bf16
    softmax_partial_kernel<<<dim3(Nn / 256, 64), 256, 0, stream>>>(Tf, pmax,
                                                                   psum);
    softmax_combine_kernel<<<Nn / 256, 256, 0, stream>>>(pmax, psum, cmx, cin);
    softmax_norm_kernel<<<(Nn * Nn) / 2048, 256, 0, stream>>>(Tf, cmx, cin,
                                                              Tsm);
    // O[n,c] = sum_m Tsm[n,m]*Vt[c,m]; fused residual out = x + 0.1*O (flat)
    gemm_bf16_wmma<64, 128, 2><<<dim3(Cc / 128, Nn / 64, 1), 256, 0, stream>>>(
        Tsm, Nn, 0, Vtb, Nn, 0, Nn, nullptr, nullptr, Cc, 0, nullptr,
        x + (size_t)b * Cc * Nn, out + (size_t)b * Cc * Nn);
  }
}
